// Attention_65859028517129
// MI455X (gfx1250) — matrix operand order repaired and checked by reading, not compiled, not run on hardware
//
#include <hip/hip_runtime.h>
#include <hip/hip_bf16.h>

// ---------------------------------------------------------------------------
// Types for CDNA5 WMMA (wave32, 16x16x32 bf16 -> f32 accumulate)
// ---------------------------------------------------------------------------
typedef __bf16 bf16_t;
typedef __attribute__((ext_vector_type(16))) __bf16 v16bf;
typedef __attribute__((ext_vector_type(8)))  __bf16 v8bf;
typedef __attribute__((ext_vector_type(8)))  float  v8f;
typedef __attribute__((ext_vector_type(4)))  float  v4f;

union FragU { v16bf v; v8bf h[2]; };

// A-matrix fragment (16x32 bf16) from an LDS row-major [M][K] tile.
// Lane L (0-15) holds row M=L: K=0..15 contiguous
// Lane L+16 holds row M=L:    K=16..31 contiguous
__device__ __forceinline__ v16bf frag_a_lds(const bf16_t* rowp, int half) {
    return *(const v16bf*)(rowp + half * 16);
}

// B-matrix fragment (32x16 bf16) from an LDS [N][K] tile (row = n, K contig).
// Lane n (0-15): K=0..15 contiguous; lane n+16: K=16..31 contiguous.
__device__ __forceinline__ v16bf frag_b_lds(const bf16_t* rowp, int half) {
    return *(const v16bf*)(rowp + half * 16);
}

__device__ __forceinline__ v8f wmma_bf16(v16bf a, v16bf b, v8f c) {
    // (neg_a, A, neg_b, B, c_mod, C, reuse_a, reuse_b)
    return __builtin_amdgcn_wmma_f32_16x16x32_bf16(false, a, false, b,
                                                   (short)0, c, false, false);
}

// ---------------------------------------------------------------------------
// gfx1250 async global->LDS copy (16B per lane), tracked by ASYNCcnt.
// VDST operand is the LDS byte address (low 32 bits of the generic pointer:
// the shared-memory aperture lives entirely in addr[63:32]).
// ---------------------------------------------------------------------------
__device__ __forceinline__ void async_copy16(const bf16_t* gsrc, bf16_t* ldst) {
    unsigned lds_addr = (unsigned)(size_t)ldst;
    unsigned long long gaddr = (unsigned long long)(size_t)gsrc;
    asm volatile("global_load_async_to_lds_b128 %0, %1, off"
                 :: "v"(lds_addr), "v"(gaddr)
                 : "memory");
}
__device__ __forceinline__ void wait_async0() {
    asm volatile("s_wait_asynccnt 0" ::: "memory");
}

// ---------------------------------------------------------------------------
// Kernel 1: projection GEMM  Out = x @ W^T + bias,  stored head-major bf16
//   x: [16384, 512] f32, W: [512, 512] f32 row-major, Out: [B,H,N,64] bf16
// Block: 256 thr (8 waves), tile 128(M) x 64(N), K-step 32.
// (f32->bf16 conversion forces the VGPR staging path here.)
// ---------------------------------------------------------------------------
__global__ __launch_bounds__(256)
void k_proj(const float* __restrict__ X, const float* __restrict__ W,
            const float* __restrict__ bias, bf16_t* __restrict__ Out) {
    __shared__ __align__(128) bf16_t As[128][32];
    __shared__ __align__(128) bf16_t Bs[64][32];
    const int t = threadIdx.x;
    const int wave = t >> 5, lane = t & 31, laneLo = lane & 15, half = lane >> 4;
    const int wm = wave >> 1, wn = wave & 1;
    const int M0 = blockIdx.x * 128;
    const int N0 = blockIdx.y * 64;

    v8f acc[2][2] = {};
    for (int k0 = 0; k0 < 512; k0 += 32) {
        // A tile: 128x32 f32 -> bf16 (1024 float4 chunks, 4 per thread)
#pragma unroll
        for (int i = 0; i < 4; ++i) {
            int idx = t + i * 256;
            int r = idx >> 3, c4 = (idx & 7) << 2;
            v4f x = *(const v4f*)(X + (size_t)(M0 + r) * 512 + k0 + c4);
            bf16_t* dst = &As[r][c4];
            dst[0] = (bf16_t)x.x; dst[1] = (bf16_t)x.y;
            dst[2] = (bf16_t)x.z; dst[3] = (bf16_t)x.w;
        }
        // B tile: 64x32, Bs[n][k] = W[N0+n][k0+k]  ([N][K] layout == W rows)
#pragma unroll
        for (int i = 0; i < 2; ++i) {
            int idx = t + i * 256;
            int r = idx >> 3, c4 = (idx & 7) << 2;
            v4f x = *(const v4f*)(W + (size_t)(N0 + r) * 512 + k0 + c4);
            bf16_t* dst = &Bs[r][c4];
            dst[0] = (bf16_t)x.x; dst[1] = (bf16_t)x.y;
            dst[2] = (bf16_t)x.z; dst[3] = (bf16_t)x.w;
        }
        __syncthreads();
        v16bf afr[2], bfr[2];
#pragma unroll
        for (int i = 0; i < 2; ++i)
            afr[i] = frag_a_lds(&As[wm * 32 + i * 16 + laneLo][0], half);
#pragma unroll
        for (int j = 0; j < 2; ++j)
            bfr[j] = frag_b_lds(&Bs[wn * 32 + j * 16 + laneLo][0], half);
#pragma unroll
        for (int i = 0; i < 2; ++i)
#pragma unroll
            for (int j = 0; j < 2; ++j)
                acc[i][j] = wmma_bf16(afr[i], bfr[j], acc[i][j]);
        __syncthreads();
    }
    // Epilogue: bias add, store head-major [B,H,N,Dh] bf16
#pragma unroll
    for (int i = 0; i < 2; ++i) {
#pragma unroll
        for (int j = 0; j < 2; ++j) {
            int n = N0 + wn * 32 + j * 16 + laneLo;
            float bvv = bias[n];
            int h = n >> 6, dh = n & 63;
#pragma unroll
            for (int r = 0; r < 8; ++r) {
                int m = M0 + wm * 32 + i * 16 + r + half * 8;
                int b = m >> 10, seq = m & 1023;
                size_t oidx = ((size_t)((b * 8 + h) * 1024 + seq)) * 64 + dh;
                Out[oidx] = (bf16_t)(acc[i][j][r] + bvv);
            }
        }
    }
}

// ---------------------------------------------------------------------------
// Kernel 2: streaming attention with additive+multiplicative distance bias.
//   S = (Qh Kh^T)*scale + d ; P = softmax(S) ; O = (P .* d) @ Vh
// Online softmax: l = running sum of exp(s-m) (d-free), numerator accumulates
// exp(s-m)*d*v; max-rescaling as usual.
// Q and K tiles stream global->LDS via ASYNC loads (no VGPR round trip);
// V needs an on-the-fly transpose so it keeps the VGPR path.
// ---------------------------------------------------------------------------
__global__ __launch_bounds__(256)
void k_attn(const bf16_t* __restrict__ Qh, const bf16_t* __restrict__ Kh,
            const bf16_t* __restrict__ Vh, const float* __restrict__ D,
            bf16_t* __restrict__ Xout, float scale) {
    __shared__ __align__(128) bf16_t Qs[128][64];
    __shared__ __align__(128) bf16_t Ks[64][64];   // [key][dh]
    __shared__ __align__(128) bf16_t Vts[64][64];  // [dh][key]  (transposed)
    __shared__ __align__(128) bf16_t Ps[8][16][64]; // per-wave P staging
    const int t = threadIdx.x;
    const int wave = t >> 5, lane = t & 31, laneLo = lane & 15, half = lane >> 4;
    const int bh = blockIdx.y;            // b*8 + h
    const int bb = bh >> 3, hh = bh & 7;
    const int q0 = blockIdx.x * 128;
    const size_t headBase = (size_t)bh * 1024 * 64;

    // Load Q tile (128x64 bf16 = 1024 chunks of 16B) -- async to LDS
#pragma unroll
    for (int i = 0; i < 4; ++i) {
        int c = t + i * 256;
        int r = c >> 3, off = (c & 7) << 3;
        async_copy16(Qh + headBase + (size_t)(q0 + r) * 64 + off, &Qs[r][off]);
    }
    wait_async0();
    __syncthreads();
    v16bf qf[2];
#pragma unroll
    for (int kh = 0; kh < 2; ++kh)
        qf[kh] = frag_a_lds(&Qs[wave * 16 + laneLo][kh * 32], half);

    v8f o[4] = {};
    float mrow[8], lrow[8];
#pragma unroll
    for (int r = 0; r < 8; ++r) { mrow[r] = -1e30f; lrow[r] = 0.0f; }
    const float* Drow = D + (size_t)bb * 1024 * 1024;

    for (int kb = 0; kb < 1024; kb += 64) {
        __syncthreads();
        // K tile -> Ks[key][dh]  (straight copy: async to LDS)
#pragma unroll
        for (int i = 0; i < 2; ++i) {
            int c = t + i * 256;
            int r = c >> 3, off = (c & 7) << 3;
            async_copy16(Kh + headBase + (size_t)(kb + r) * 64 + off,
                         &Ks[r][off]);
        }
        // V tile -> Vts[dh][key] (transpose on LDS store)
#pragma unroll
        for (int i = 0; i < 2; ++i) {
            int c = t + i * 256;
            int r = c >> 3, off = (c & 7) << 3;
            v8bf vv = *(const v8bf*)(Vh + headBase + (size_t)(kb + r) * 64 + off);
#pragma unroll
            for (int e = 0; e < 8; ++e) Vts[off + e][r] = vv[e];
        }
        wait_async0();
        __syncthreads();

        // Scores: 4 key tiles of 16, Dh=64 -> 2 WMMA each
        float sv[4][8], dv[4][8];
#pragma unroll
        for (int j = 0; j < 4; ++j) {
            v8f s = {};
            s = wmma_bf16(qf[0], frag_b_lds(&Ks[j * 16 + laneLo][0], half), s);
            s = wmma_bf16(qf[1], frag_b_lds(&Ks[j * 16 + laneLo][32], half), s);
            int key = kb + j * 16 + laneLo;
#pragma unroll
            for (int r = 0; r < 8; ++r) {
                int qrow = q0 + wave * 16 + r + half * 8;
                // d is streamed once per pass: non-temporal, keep L2 for Q/K/V
                float dval =
                    __builtin_nontemporal_load(Drow + (size_t)qrow * 1024 + key);
                dv[j][r] = dval;
                sv[j][r] = s[r] * scale + dval;
            }
        }
        // Online softmax (rows live in one 16-lane half each)
#pragma unroll
        for (int r = 0; r < 8; ++r) {
            float mt = fmaxf(fmaxf(sv[0][r], sv[1][r]), fmaxf(sv[2][r], sv[3][r]));
#pragma unroll
            for (int off = 8; off >= 1; off >>= 1)
                mt = fmaxf(mt, __shfl_xor(mt, off, 32));
            float mnew = fmaxf(mrow[r], mt);
            float resc = __expf(mrow[r] - mnew);
            mrow[r] = mnew;
            lrow[r] *= resc;
#pragma unroll
            for (int tt = 0; tt < 4; ++tt) o[tt][r] *= resc;
            float ps = 0.0f;
#pragma unroll
            for (int j = 0; j < 4; ++j) {
                float p = __expf(sv[j][r] - mnew);
                ps += p;
                sv[j][r] = p * dv[j][r];   // post-softmax multiplicative bias
            }
#pragma unroll
            for (int off = 8; off >= 1; off >>= 1)
                ps += __shfl_xor(ps, off, 32);
            lrow[r] += ps;
        }
        // Stage P (C-layout -> row-major) into per-wave LDS scratch
#pragma unroll
        for (int j = 0; j < 4; ++j)
#pragma unroll
            for (int r = 0; r < 8; ++r)
                Ps[wave][r + half * 8][j * 16 + laneLo] = (bf16_t)sv[j][r];
        asm volatile("s_wait_dscnt 0" ::: "memory"); // wave-local LDS RAW fence
        // O += P @ V  (A = P[16x32 keys], B from Vts [dh][key])
#pragma unroll
        for (int kh = 0; kh < 2; ++kh) {
            v16bf pf = frag_a_lds(&Ps[wave][laneLo][kh * 32], half);
#pragma unroll
            for (int tt = 0; tt < 4; ++tt)
                o[tt] = wmma_bf16(pf,
                    frag_b_lds(&Vts[tt * 16 + laneLo][kh * 32], half), o[tt]);
        }
    }
    // Epilogue: normalize, store bf16 to [B,N,E] (heads re-interleaved)
#pragma unroll
    for (int tt = 0; tt < 4; ++tt) {
        int dh = tt * 16 + laneLo;
#pragma unroll
        for (int r = 0; r < 8; ++r) {
            int qrow = q0 + wave * 16 + r + half * 8;
            float inv = 1.0f / lrow[r];
            size_t oidx = ((size_t)(bb * 1024 + qrow)) * 512 + hh * 64 + dh;
            Xout[oidx] = (bf16_t)(o[tt][r] * inv);
        }
    }
}

// ---------------------------------------------------------------------------
// Kernel 3: output projection  out = Xw(bf16) @ Wp^T + bp,  f32 output
// A tile is already bf16 -> async global->LDS copy; W needs conversion.
// ---------------------------------------------------------------------------
__global__ __launch_bounds__(256)
void k_outproj(const bf16_t* __restrict__ X, const float* __restrict__ W,
               const float* __restrict__ bias, float* __restrict__ Out) {
    __shared__ __align__(128) bf16_t As[128][32];
    __shared__ __align__(128) bf16_t Bs[64][32];
    const int t = threadIdx.x;
    const int wave = t >> 5, lane = t & 31, laneLo = lane & 15, half = lane >> 4;
    const int wm = wave >> 1, wn = wave & 1;
    const int M0 = blockIdx.x * 128;
    const int N0 = blockIdx.y * 64;

    v8f acc[2][2] = {};
    for (int k0 = 0; k0 < 512; k0 += 32) {
        // A tile: bf16 source (512 chunks of 16B, 2/thread) -- async to LDS
#pragma unroll
        for (int i = 0; i < 2; ++i) {
            int c = t + i * 256;
            int r = c >> 2, off = (c & 3) << 3;
            async_copy16(X + (size_t)(M0 + r) * 512 + k0 + off, &As[r][off]);
        }
        // B tile from f32 W rows (convert in VGPRs)
#pragma unroll
        for (int i = 0; i < 2; ++i) {
            int idx = t + i * 256;
            int r = idx >> 3, c4 = (idx & 7) << 2;
            v4f x = *(const v4f*)(W + (size_t)(N0 + r) * 512 + k0 + c4);
            bf16_t* dst = &Bs[r][c4];
            dst[0] = (bf16_t)x.x; dst[1] = (bf16_t)x.y;
            dst[2] = (bf16_t)x.z; dst[3] = (bf16_t)x.w;
        }
        wait_async0();
        __syncthreads();
        v16bf afr[2], bfr[2];
#pragma unroll
        for (int i = 0; i < 2; ++i)
            afr[i] = frag_a_lds(&As[wm * 32 + i * 16 + laneLo][0], half);
#pragma unroll
        for (int j = 0; j < 2; ++j)
            bfr[j] = frag_b_lds(&Bs[wn * 32 + j * 16 + laneLo][0], half);
#pragma unroll
        for (int i = 0; i < 2; ++i)
#pragma unroll
            for (int j = 0; j < 2; ++j)
                acc[i][j] = wmma_bf16(afr[i], bfr[j], acc[i][j]);
        __syncthreads();
    }
#pragma unroll
    for (int i = 0; i < 2; ++i) {
#pragma unroll
        for (int j = 0; j < 2; ++j) {
            int n = N0 + wn * 32 + j * 16 + laneLo;
            float bvv = bias[n];
#pragma unroll
            for (int r = 0; r < 8; ++r) {
                int m = M0 + wm * 32 + i * 16 + r + half * 8;
                // final output is write-only: non-temporal store
                __builtin_nontemporal_store(acc[i][j][r] + bvv,
                                            Out + (size_t)m * 512 + n);
            }
        }
    }
}

// ---------------------------------------------------------------------------
// Host launcher
// ---------------------------------------------------------------------------
extern "C" void kernel_launch(void* const* d_in, const int* in_sizes, int n_in,
                              void* d_out, int out_size, void* d_ws, size_t ws_size,
                              hipStream_t stream) {
    const float* q  = (const float*)d_in[0];
    const float* k  = (const float*)d_in[1];
    const float* v  = (const float*)d_in[2];
    const float* d  = (const float*)d_in[3];
    const float* Wq = (const float*)d_in[4];
    const float* bq = (const float*)d_in[5];
    const float* Wk = (const float*)d_in[6];
    const float* bk = (const float*)d_in[7];
    const float* Wv = (const float*)d_in[8];
    const float* bv = (const float*)d_in[9];
    const float* Wp = (const float*)d_in[10];
    const float* bp = (const float*)d_in[11];
    float* out = (float*)d_out;

    const size_t headElems = (size_t)16 * 8 * 1024 * 64; // 8.39M per tensor
    bf16_t* Qw = (bf16_t*)d_ws;
    bf16_t* Kw = Qw + headElems;
    bf16_t* Vw = Kw + headElems;
    bf16_t* Xw = Vw + headElems; // [16384, 512] bf16

    dim3 blk(256);
    dim3 g1(128, 8);   // M=16384/128, N=512/64
    k_proj<<<g1, blk, 0, stream>>>(q, Wq, bq, Qw);
    k_proj<<<g1, blk, 0, stream>>>(k, Wk, bk, Kw);
    k_proj<<<g1, blk, 0, stream>>>(v, Wv, bv, Vw);

    dim3 g2(8, 128);   // 8 query tiles x (B*H)
    k_attn<<<g2, blk, 0, stream>>>(Qw, Kw, Vw, d, Xw, 0.125f);

    k_outproj<<<g1, blk, 0, stream>>>(Xw, Wp, bp, out);
}
